// MixtureOfDepths_61821759259058
// MI455X (gfx1250) — matrix operand order repaired and checked by reading, not compiled, not run on hardware
//
#include <hip/hip_runtime.h>
#include <hip/hip_bf16.h>

typedef __bf16 bf16_t;
typedef __attribute__((ext_vector_type(16))) __bf16 v16bf;
typedef __attribute__((ext_vector_type(8)))  float  v8f;
typedef __attribute__((ext_vector_type(4)))  float  f32x4;

union FragAB { f32x4 f[2]; v16bf v; };
union Pack8  { f32x4 f; bf16_t h[8]; };

#define HDIM 1024
#define SDIM 8192
#define BDIM 8
#define NROUTE 256
#define CAP 4096

__device__ __forceinline__ f32x4 cvt8_bf16(f32x4 a, f32x4 b) {
  Pack8 p;
  p.h[0] = (bf16_t)a.x; p.h[1] = (bf16_t)a.y; p.h[2] = (bf16_t)a.z; p.h[3] = (bf16_t)a.w;
  p.h[4] = (bf16_t)b.x; p.h[5] = (bf16_t)b.y; p.h[6] = (bf16_t)b.z; p.h[7] = (bf16_t)b.w;
  return p.f;
}

// Async copy of one 16B chunk: global (bf16, already in target layout) -> LDS.
// Uses the CDNA5 async-to-LDS path (tracked by ASYNCcnt), bypassing VGPRs.
__device__ __forceinline__ void async_b128(const bf16_t* g, bf16_t* l) {
  unsigned lds  = (unsigned)(size_t)l;              // flat->LDS: addr[31:0]
  unsigned long long ga = (unsigned long long)(size_t)g;
  asm volatile("global_load_async_to_lds_b128 %0, %1, off"
               :: "v"(lds), "v"(ga) : "memory");
}
__device__ __forceinline__ void wait_async0() {
  asm volatile("s_wait_asynccnt 0x0" ::: "memory");
}

// ---------------------------------------------------------------------------
// Kernel 1: f32 [R][C] -> bf16 [C][R] (transpose + convert). Weights are tiny.
// ---------------------------------------------------------------------------
__global__ void mod_cvt_transpose(const float* __restrict__ src,
                                  bf16_t* __restrict__ dst, int R, int C) {
  int idx = blockIdx.x * blockDim.x + threadIdx.x;
  if (idx >= R * C) return;
  int r = idx / C, c = idx - r * C;
  dst[(size_t)c * R + r] = (bf16_t)src[(size_t)r * C + c];
}

// ---------------------------------------------------------------------------
// Kernel 2: router.  scores = relu(x@W1 + b1)@W2 + b2 ; rweights = sigmoid.
// Block tile: 64 tokens x 256 dims, 8 waves (2M x 4N), wave tile 32x64.
// ---------------------------------------------------------------------------
__global__ __launch_bounds__(256) void mod_router(
    const float* __restrict__ x,       // [B*S][H]
    const bf16_t* __restrict__ W1t,    // [256][1024] n-major
    const float* __restrict__ b1,      // [256]
    const float* __restrict__ W2,      // [256]
    const float* __restrict__ b2,      // [1]
    float* __restrict__ scores,        // [B*S]
    float* __restrict__ rweights) {    // [B*S]
  __shared__ __align__(16) bf16_t As[64][40];
  __shared__ __align__(16) bf16_t Bs[NROUTE][40];
  __shared__ float b1s[NROUTE];
  __shared__ float w2s[NROUTE];
  __shared__ float sbuf[64];

  const int tid  = threadIdx.x;
  const int lane = tid & 31;
  const int wave = tid >> 5;
  const int t0   = blockIdx.x * 64;

  if (tid < NROUTE) { b1s[tid] = b1[tid]; w2s[tid] = W2[tid]; }
  if (tid < 64) sbuf[tid] = b2[0];

  const int m_wave = (wave >> 2) * 32;   // 0 or 32
  const int n_wave = (wave & 3) * 64;    // 0..192

  v8f acc[2][4] = {};

  for (int k0 = 0; k0 < HDIM; k0 += 32) {
    __syncthreads();
    { // B tile: 256 rows x 32 bf16; one thread = one row = 4x16B async chunks
      const bf16_t* s = W1t + (size_t)tid * HDIM + k0;
      async_b128(s,      &Bs[tid][0]);
      async_b128(s + 8,  &Bs[tid][8]);
      async_b128(s + 16, &Bs[tid][16]);
      async_b128(s + 24, &Bs[tid][24]);
    }
    { // A tile: 64 rows x 32 f32 -> bf16; one thread = 8 floats
      int row = tid >> 2;
      int seg = (tid & 3) * 8;
      const float* s = x + (size_t)(t0 + row) * HDIM + k0 + seg;
      f32x4 v0 = *(const f32x4*)(s);
      f32x4 v1 = *(const f32x4*)(s + 4);
      *(f32x4*)&As[row][seg] = cvt8_bf16(v0, v1);
    }
    wait_async0();
    __syncthreads();

    FragAB a[2], b[4];
    const int kb  = (lane >> 4) * 8;   // A: lane-half selects K octet
    const int kb2 = (lane >> 4) * 8;   // B: lane-half selects K octet
#pragma unroll
    for (int i = 0; i < 2; i++) {
      int arow = m_wave + i * 16 + (lane & 15);
      a[i].f[0] = *(const f32x4*)&As[arow][kb];
      a[i].f[1] = *(const f32x4*)&As[arow][kb + 16];
    }
#pragma unroll
    for (int j = 0; j < 4; j++) {
      int brow = n_wave + j * 16 + (lane & 15);
      b[j].f[0] = *(const f32x4*)&Bs[brow][kb2];
      b[j].f[1] = *(const f32x4*)&Bs[brow][kb2 + 16];
    }
#pragma unroll
    for (int i = 0; i < 2; i++)
#pragma unroll
      for (int j = 0; j < 4; j++)
        acc[i][j] = __builtin_amdgcn_wmma_f32_16x16x32_bf16(
            false, a[i].v, false, b[j].v, (short)0, acc[i][j], false, false);
  }

  // epilogue: relu(+b1) * W2, reduce over d via LDS float atomics
#pragma unroll
  for (int i = 0; i < 2; i++) {
#pragma unroll
    for (int r = 0; r < 8; r++) {
      int mloc = m_wave + i * 16 + ((lane >> 4) * 8) + r;
      float part = 0.f;
#pragma unroll
      for (int j = 0; j < 4; j++) {
        int d = n_wave + j * 16 + (lane & 15);
        float h = acc[i][j][r] + b1s[d];
        h = h > 0.f ? h : 0.f;
        part += h * w2s[d];
      }
      atomicAdd(&sbuf[mloc], part);
    }
  }
  __syncthreads();
  if (tid < 64) {
    float s = sbuf[tid];
    scores[t0 + tid]   = s;
    rweights[t0 + tid] = 1.f / (1.f + __expf(-s));
  }
}

// ---------------------------------------------------------------------------
// Kernel 3: per-row top-4096 of 8192 via in-LDS bitonic sort of packed u64
// (monotonic float bits << 32 | ~index) -> matches top_k tie-breaking.
// ---------------------------------------------------------------------------
__global__ __launch_bounds__(1024) void mod_topk(const float* __restrict__ scores,
                                                 int* __restrict__ sel) {
  extern __shared__ unsigned long long keys[];  // 8192 * 8B = 64 KB
  const int b   = blockIdx.x;
  const int tid = threadIdx.x;
  for (int i = tid; i < SDIM; i += 1024) {
    unsigned u = __float_as_uint(scores[b * SDIM + i]);
    u = (u & 0x80000000u) ? ~u : (u | 0x80000000u);
    keys[i] = ((unsigned long long)u << 32) | (unsigned)(~i);
  }
  for (unsigned k = 2; k <= SDIM; k <<= 1) {
    for (unsigned j = k >> 1; j > 0; j >>= 1) {
      __syncthreads();
      for (unsigned i = tid; i < SDIM; i += 1024) {
        unsigned ixj = i ^ j;
        if (ixj > i) {
          unsigned long long a = keys[i], c = keys[ixj];
          bool asc = ((i & k) == 0);
          if (asc ? (a > c) : (a < c)) { keys[i] = c; keys[ixj] = a; }
        }
      }
    }
  }
  __syncthreads();
  for (int t = tid; t < CAP; t += 1024) {  // top half of ascending sort
    unsigned long long key = keys[CAP + t];
    sel[b * CAP + t] = (int)(~(unsigned)(key & 0xffffffffu));
  }
}

// ---------------------------------------------------------------------------
// Kernel 4: bulk copy x -> out (selected rows overwritten by kernel 5).
// ---------------------------------------------------------------------------
__global__ __launch_bounds__(256) void mod_copy(const f32x4* __restrict__ src,
                                                f32x4* __restrict__ dst) {
  size_t base = (size_t)blockIdx.x * 1024 + threadIdx.x;
  dst[base]       = src[base];
  dst[base + 256] = src[base + 256];
  dst[base + 512] = src[base + 512];
  dst[base + 768] = src[base + 768];
}

// ---------------------------------------------------------------------------
// Kernel 5: gathered GEMM + GELU + scatter.
// Block tile 128(M) x 64(N), 8 waves (4M x 2N), wave tile 32x32, K-step 32.
// ---------------------------------------------------------------------------
__global__ __launch_bounds__(256) void mod_moe_gemm(
    const float* __restrict__ x,      // [B][S][H]
    const bf16_t* __restrict__ Wlt,   // [1024][1024] n-major
    const int* __restrict__ sel,      // [B][CAP]
    float* __restrict__ out) {        // [B][S][H]
  __shared__ __align__(16) bf16_t As[128][40];
  __shared__ __align__(16) bf16_t Bs[64][40];
  __shared__ int toks[128];

  const int tid  = threadIdx.x;
  const int lane = tid & 31;
  const int wave = tid >> 5;
  const int n0   = blockIdx.x * 64;
  const int m0   = blockIdx.y * 128;
  const int bb   = blockIdx.z;

  if (tid < 128) toks[tid] = sel[bb * CAP + m0 + tid];

  const int m_wave = (wave >> 1) * 32;  // 0..96
  const int n_wave = (wave & 1) * 32;   // 0 or 32

  v8f acc[2][2] = {};
  const float* xb = x + (size_t)bb * SDIM * HDIM;

  for (int k0 = 0; k0 < HDIM; k0 += 32) {
    __syncthreads();
    { // B tile: 64 rows x 32 bf16 = 4KB; one 16B async chunk per thread
      int row = tid >> 2;
      int seg = (tid & 3) * 8;
      async_b128(Wlt + (size_t)(n0 + row) * HDIM + k0 + seg, &Bs[row][seg]);
    }
    { // A tile: 128 gathered rows x 32 f32 -> bf16; one thread = 16 floats
      int row = tid >> 1;
      int cs  = (tid & 1) * 16;
      const float* s = xb + (size_t)toks[row] * HDIM + k0 + cs;
      f32x4 v0 = *(const f32x4*)(s);
      f32x4 v1 = *(const f32x4*)(s + 4);
      f32x4 v2 = *(const f32x4*)(s + 8);
      f32x4 v3 = *(const f32x4*)(s + 12);
      *(f32x4*)&As[row][cs]     = cvt8_bf16(v0, v1);
      *(f32x4*)&As[row][cs + 8] = cvt8_bf16(v2, v3);
    }
    wait_async0();
    __syncthreads();

    FragAB a[2], b[2];
    const int kb  = (lane >> 4) * 8;
    const int kb2 = (lane >> 4) * 8;
#pragma unroll
    for (int i = 0; i < 2; i++) {
      int arow = m_wave + i * 16 + (lane & 15);
      a[i].f[0] = *(const f32x4*)&As[arow][kb];
      a[i].f[1] = *(const f32x4*)&As[arow][kb + 16];
    }
#pragma unroll
    for (int j = 0; j < 2; j++) {
      int brow = n_wave + j * 16 + (lane & 15);
      b[j].f[0] = *(const f32x4*)&Bs[brow][kb2];
      b[j].f[1] = *(const f32x4*)&Bs[brow][kb2 + 16];
    }
#pragma unroll
    for (int i = 0; i < 2; i++)
#pragma unroll
      for (int j = 0; j < 2; j++)
        acc[i][j] = __builtin_amdgcn_wmma_f32_16x16x32_bf16(
            false, a[i].v, false, b[j].v, (short)0, acc[i][j], false, false);
  }

  // epilogue: tanh-GELU + scatter to out[b, tok, n]
  float* ob = out + (size_t)bb * SDIM * HDIM;
#pragma unroll
  for (int i = 0; i < 2; i++) {
    int mbase = m_wave + i * 16 + ((lane >> 4) * 8);
#pragma unroll
    for (int j = 0; j < 2; j++) {
      int n = n0 + n_wave + j * 16 + (lane & 15);
#pragma unroll
      for (int r = 0; r < 8; r++) {
        float v = acc[i][j][r];
        float g = 0.5f * v *
                  (1.f + tanhf(0.7978845608028654f * (v + 0.044715f * v * v * v)));
        ob[(size_t)toks[mbase + r] * HDIM + n] = g;
      }
    }
  }
}

// ---------------------------------------------------------------------------
extern "C" void kernel_launch(void* const* d_in, const int* in_sizes, int n_in,
                              void* d_out, int out_size, void* d_ws, size_t ws_size,
                              hipStream_t stream) {
  const float* x  = (const float*)d_in[0];   // [8][8192][1024]
  const float* W1 = (const float*)d_in[1];   // [1024][256]
  const float* b1 = (const float*)d_in[2];   // [256]
  const float* W2 = (const float*)d_in[3];   // [256][1] -> [256]
  const float* b2 = (const float*)d_in[4];   // [1]
  const float* Wl = (const float*)d_in[5];   // [1024][1024]

  float* out_main = (float*)d_out;                          // [8][8192][1024]
  float* out_rw   = out_main + (size_t)BDIM * SDIM * HDIM;  // [8][8192][1]

  char* ws = (char*)d_ws;
  float*  scores = (float*)ws;                              // 256 KB
  bf16_t* W1t    = (bf16_t*)(ws + 262144);                  // 512 KB
  bf16_t* Wlt    = (bf16_t*)(ws + 262144 + 524288);         // 2 MB
  int*    sel    = (int*)(ws + 262144 + 524288 + 2097152);  // 128 KB

  mod_cvt_transpose<<<(1024 * 256 + 255) / 256, 256, 0, stream>>>(W1, W1t, 1024, 256);
  mod_cvt_transpose<<<(1024 * 1024 + 255) / 256, 256, 0, stream>>>(Wl, Wlt, 1024, 1024);

  mod_router<<<(BDIM * SDIM) / 64, 256, 0, stream>>>(x, W1t, b1, W2, b2, scores, out_rw);

  mod_topk<<<BDIM, 1024, SDIM * sizeof(unsigned long long), stream>>>(scores, sel);

  mod_copy<<<(BDIM * SDIM * HDIM) / 4096, 256, 0, stream>>>((const f32x4*)x,
                                                            (f32x4*)out_main);

  mod_moe_gemm<<<dim3(HDIM / 64, CAP / 128, BDIM), 256, 0, stream>>>(x, Wlt, sel,
                                                                     out_main);
}
